// SimpleMHA_41360535061210
// MI455X (gfx1250) — matrix operand order repaired and checked by reading, not compiled, not run on hardware
//
#include <hip/hip_runtime.h>
#include <hip/hip_bf16.h>

// ---------------------------------------------------------------------------
// MHA forward for gfx1250 (MI455X): f16 WMMA (v_wmma_f32_16x16x32_f16),
// two-pass flash attention, A materialized exactly once (write-bound: 268 MB).
// ---------------------------------------------------------------------------

typedef __attribute__((ext_vector_type(16))) _Float16 v16h;
typedef __attribute__((ext_vector_type(8)))  _Float16 v8h;
typedef __attribute__((ext_vector_type(8)))  float    v8f;
typedef __attribute__((ext_vector_type(4)))  float    v4f;

constexpr int kB  = 2;
constexpr int kT  = 2048;
constexpr int kD  = 512;
constexpr int kH  = 8;
constexpr int kDK = 64;

__device__ __forceinline__ v8f wmma16(v16h a, v16h b, v8f c) {
  // D = A(16x32 f16) * B(32x16 f16) + C(16x16 f32)
  return __builtin_amdgcn_wmma_f32_16x16x32_f16(
      /*neg_a=*/false, a, /*neg_b=*/false, b,
      /*c_mod=*/(short)0, c, /*reuse_a=*/false, /*reuse_b=*/false);
}

__device__ __forceinline__ v16h cat8(v8h a, v8h b) {
  return __builtin_shufflevector(a, b, 0, 1, 2, 3, 4, 5, 6, 7,
                                       8, 9, 10, 11, 12, 13, 14, 15);
}

__device__ __forceinline__ v16h cvt16(v4f a, v4f b, v4f c, v4f d) {
  v16h r;
  r[0]  = (_Float16)a[0]; r[1]  = (_Float16)a[1];
  r[2]  = (_Float16)a[2]; r[3]  = (_Float16)a[3];
  r[4]  = (_Float16)b[0]; r[5]  = (_Float16)b[1];
  r[6]  = (_Float16)b[2]; r[7]  = (_Float16)b[3];
  r[8]  = (_Float16)c[0]; r[9]  = (_Float16)c[1];
  r[10] = (_Float16)c[2]; r[11] = (_Float16)c[3];
  r[12] = (_Float16)d[0]; r[13] = (_Float16)d[1];
  r[14] = (_Float16)d[2]; r[15] = (_Float16)d[3];
  return r;
}

// B-fragment (32x16, f16) built from 16 contiguous fp32 (one W row chunk).
__device__ __forceinline__ v16h bfrag_from_f32(const float* p) {
  return cvt16(*(const v4f*)(p), *(const v4f*)(p + 4),
               *(const v4f*)(p + 8), *(const v4f*)(p + 12));
}

// ---------------------------------------------------------------------------
// Projection: Yh = f16( X @ W^T + bias ), scattered into head layouts.
//   mode 0: Q -> [B,H,T,DK] f16, scaled by 1/sqrt(DK)=0.125 (exact in f16)
//   mode 1: K -> [B,H,T,DK] f16
//   mode 2: V -> [B,H,DK,T] f16 (transposed, for contiguous B-fragments)
// One wave computes a 16-row x 64-col strip (4 accumulator tiles).
// ---------------------------------------------------------------------------
__global__ __launch_bounds__(128) void mha_proj(const float* __restrict__ X,
                                                const float* __restrict__ W,
                                                const float* __restrict__ bias,
                                                _Float16* __restrict__ Yh,
                                                int mode) {
  const int lane = threadIdx.x;
  const int hl = lane >> 4;      // lane half (0/1)
  const int ln = lane & 15;
  const int gwid = blockIdx.x * 4 + threadIdx.y;
  const int mtile = gwid >> 3;   // 0..255 over B*T/16 rows
  const int nstrip = gwid & 7;   // 0..7, 64 output cols each
  const int rowbase = mtile * 16;

  v8f acc[4] = {{}, {}, {}, {}};
  const float* xrow = X + (size_t)(rowbase + ln) * kD;

  for (int c = 0; c < kD; c += 32) {
    // A fragment: row ln, K pattern {16*hl..16*hl+15}
    v4f x0 = *(const v4f*)(xrow + c + 16 * hl);
    v4f x1 = *(const v4f*)(xrow + c + 16 * hl + 4);
    v4f x2 = *(const v4f*)(xrow + c + 16 * hl + 8);
    v4f x3 = *(const v4f*)(xrow + c + 16 * hl + 12);
    v16h af = cvt16(x0, x1, x2, x3);
#pragma unroll
    for (int nt = 0; nt < 4; ++nt) {
      const float* wrow =
          W + (size_t)(nstrip * 64 + nt * 16 + ln) * kD + c + 16 * hl;
      v16h bf = bfrag_from_f32(wrow);
      acc[nt] = wmma16(af, bf, acc[nt]);
    }
  }

#pragma unroll
  for (int nt = 0; nt < 4; ++nt) {
    const int ncol = nstrip * 64 + nt * 16 + ln;
    const float bv = bias[ncol];
    const int h = ncol >> 6;      // head
    const int dk = ncol & 63;
#pragma unroll
    for (int r = 0; r < 8; ++r) {
      const int row = rowbase + r + 8 * hl;   // global row in [0, B*T)
      const int b = row >> 11;                // row / kT
      const int tt = row & (kT - 1);
      float y = acc[nt][r] + bv;
      if (mode == 0) y *= 0.125f;             // bake softmax scale into Q
      size_t addr;
      if (mode == 2)
        addr = (((size_t)(b * kH + h)) * kDK + dk) * kT + tt;  // V^T
      else
        addr = (((size_t)(b * kH + h)) * kT + tt) * kDK + dk;  // Q/K
      Yh[addr] = (_Float16)y;
    }
  }
}

// ---------------------------------------------------------------------------
// Attention: per wave, 16 query rows of one (b,h). Two-pass flash:
//   pass1: exact row max/sum (online, recomputing QK^T is cheap),
//   pass2: recompute S, write normalized A once, accumulate O = P*V.
// Causal mask applied analytically (== reference's tril attn_mask).
// ---------------------------------------------------------------------------
__global__ __launch_bounds__(128) void mha_attn(const _Float16* __restrict__ Qh,
                                                const _Float16* __restrict__ Kh,
                                                const _Float16* __restrict__ VhT,
                                                _Float16* __restrict__ Oh,
                                                float* __restrict__ Aout) {
  __shared__ __align__(16) _Float16 lds[4][16][32];  // P staging per wave

  const int lane = threadIdx.x;
  const int wave = threadIdx.y;
  const int hl = lane >> 4;
  const int ln = lane & 15;
  const int gwid = blockIdx.x * 4 + wave;
  const int bh = gwid >> 7;          // b*H + h
  const int qtile = gwid & 127;
  const int qbase = qtile * 16;
  const int kend = (qbase / 32 + 1) * 32;  // covers keys <= qbase+15

  // Q fragments (held in VGPRs for the whole kernel), DK=64 -> 2 chunks
  const _Float16* qrow = Qh + ((size_t)bh * kT + qbase + ln) * kDK;
  const v16h qf0 = cat8(*(const v8h*)(qrow + 16 * hl),
                        *(const v8h*)(qrow + 16 * hl + 8));
  const v16h qf1 = cat8(*(const v8h*)(qrow + 32 + 16 * hl),
                        *(const v8h*)(qrow + 32 + 16 * hl + 8));

  const _Float16* kmat = Kh + (size_t)bh * kT * kDK;

  float rmax[8], rsum[8];
#pragma unroll
  for (int r = 0; r < 8; ++r) { rmax[r] = -1e30f; rsum[r] = 0.f; }

  // ---------------- pass 1: softmax statistics ----------------
  for (int kb = 0; kb < kend; kb += 32) {
    if (kb + 32 < kend)
      __builtin_prefetch(kmat + (size_t)(kb + 32) * kDK, 0, 1);
#pragma unroll
    for (int t = 0; t < 2; ++t) {
      const int nb = kb + t * 16;
      const _Float16* krow = kmat + (size_t)(nb + ln) * kDK;
      v16h kf0 = *(const v16h*)(krow + 16 * hl);
      v16h kf1 = *(const v16h*)(krow + 32 + 16 * hl);
      v8f s = {};
      s = wmma16(qf0, kf0, s);
      s = wmma16(qf1, kf1, s);
      const int key = nb + ln;  // this lane's column (key index)
#pragma unroll
      for (int r = 0; r < 8; ++r) {
        const int qm = qbase + r + 8 * hl;
        float sv = (key <= qm) ? s[r] : -1e30f;
        float tmx = sv;
#pragma unroll
        for (int off = 1; off < 16; off <<= 1)
          tmx = fmaxf(tmx, __shfl_xor(tmx, off, 32));
        const float nm = fmaxf(rmax[r], tmx);
        float p = __expf(sv - nm);
#pragma unroll
        for (int off = 1; off < 16; off <<= 1) p += __shfl_xor(p, off, 32);
        rsum[r] = rsum[r] * __expf(rmax[r] - nm) + p;
        rmax[r] = nm;
      }
    }
  }
  float inv[8];
#pragma unroll
  for (int r = 0; r < 8; ++r) inv[r] = 1.0f / rsum[r];  // rsum >= 1 (diag)

  // ---------------- pass 2: write A, accumulate O = P*V ----------------
  v8f oacc[4] = {{}, {}, {}, {}};
  for (int kb = 0; kb < kend; kb += 32) {
#pragma unroll
    for (int t = 0; t < 2; ++t) {
      const int nb = kb + t * 16;
      const _Float16* krow = kmat + (size_t)(nb + ln) * kDK;
      v16h kf0 = *(const v16h*)(krow + 16 * hl);
      v16h kf1 = *(const v16h*)(krow + 32 + 16 * hl);
      v8f s = {};
      s = wmma16(qf0, kf0, s);
      s = wmma16(qf1, kf1, s);
      const int key = nb + ln;
#pragma unroll
      for (int r = 0; r < 8; ++r) {
        const int qm = qbase + r + 8 * hl;
        const float p =
            (key <= qm) ? __expf(s[r] - rmax[r]) * inv[r] : 0.0f;
        Aout[((size_t)bh * kT + qm) * kT + key] = p;           // coalesced/row
        lds[wave][r + 8 * hl][t * 16 + ln] = (_Float16)p;      // C->A relayout
      }
    }
    __builtin_amdgcn_wave_barrier();  // LDS in-order within wave; keep order
    v16h pf = cat8(*(const v8h*)(&lds[wave][ln][16 * hl]),
                   *(const v8h*)(&lds[wave][ln][16 * hl + 8]));
#pragma unroll
    for (int g = 0; g < 4; ++g) {
      const _Float16* vrow =
          VhT + ((size_t)bh * kDK + g * 16 + ln) * kT + kb + 16 * hl;
      v16h vf = *(const v16h*)vrow;
      oacc[g] = wmma16(pf, vf, oacc[g]);
    }
    __builtin_amdgcn_wave_barrier();
  }

  // Zero the causally-masked tail of A (kend..T) for these 16 rows.
  {
    const v4f z = {};
    for (int r2 = 0; r2 < 16; ++r2) {
      float* row = Aout + ((size_t)bh * kT + qbase + r2) * kT;
      for (int c = kend + lane * 4; c < kT; c += 128) *(v4f*)(row + c) = z;
    }
  }

  // Store O in [B,T,D] f16 layout (heads re-interleaved) for the out-proj.
  const int b = bh >> 3, h = bh & 7;
#pragma unroll
  for (int g = 0; g < 4; ++g) {
#pragma unroll
    for (int r = 0; r < 8; ++r) {
      const int row = b * kT + qbase + r + 8 * hl;
      const int col = h * kDK + g * 16 + ln;
      Oh[(size_t)row * kD + col] = (_Float16)oacc[g][r];
    }
  }
}

// ---------------------------------------------------------------------------
// Output projection: out = Oh(f16) @ Wo^T + bo, fp32 result.
// ---------------------------------------------------------------------------
__global__ __launch_bounds__(128) void mha_outproj(
    const _Float16* __restrict__ Oh, const float* __restrict__ W,
    const float* __restrict__ bias, float* __restrict__ out) {
  const int lane = threadIdx.x;
  const int hl = lane >> 4;
  const int ln = lane & 15;
  const int gwid = blockIdx.x * 4 + threadIdx.y;
  const int mtile = gwid >> 3;
  const int nstrip = gwid & 7;
  const int rowbase = mtile * 16;

  v8f acc[4] = {{}, {}, {}, {}};
  const _Float16* orow = Oh + (size_t)(rowbase + ln) * kD;

  for (int c = 0; c < kD; c += 32) {
    v16h af = cat8(*(const v8h*)(orow + c + 16 * hl),
                   *(const v8h*)(orow + c + 16 * hl + 8));
#pragma unroll
    for (int nt = 0; nt < 4; ++nt) {
      const float* wrow =
          W + (size_t)(nstrip * 64 + nt * 16 + ln) * kD + c + 16 * hl;
      v16h bf = bfrag_from_f32(wrow);
      acc[nt] = wmma16(af, bf, acc[nt]);
    }
  }

#pragma unroll
  for (int nt = 0; nt < 4; ++nt) {
    const int ncol = nstrip * 64 + nt * 16 + ln;
    const float bv = bias[ncol];
#pragma unroll
    for (int r = 0; r < 8; ++r) {
      const int row = rowbase + r + 8 * hl;
      out[(size_t)row * kD + ncol] = acc[nt][r] + bv;
    }
  }
}

// ---------------------------------------------------------------------------
extern "C" void kernel_launch(void* const* d_in, const int* in_sizes, int n_in,
                              void* d_out, int out_size, void* d_ws,
                              size_t ws_size, hipStream_t stream) {
  (void)in_sizes; (void)n_in; (void)out_size; (void)ws_size;

  const float* q  = (const float*)d_in[0];
  const float* k  = (const float*)d_in[1];
  const float* v  = (const float*)d_in[2];
  // d_in[3] = attn_mask (causal tril) — applied analytically in mha_attn.
  const float* Wq = (const float*)d_in[4];
  const float* bq = (const float*)d_in[5];
  const float* Wk = (const float*)d_in[6];
  const float* bk = (const float*)d_in[7];
  const float* Wv = (const float*)d_in[8];
  const float* bv = (const float*)d_in[9];
  const float* Wo = (const float*)d_in[10];
  const float* bo = (const float*)d_in[11];

  float* out  = (float*)d_out;                      // [B,T,D]
  float* Aout = out + (size_t)kB * kT * kD;         // [B,H,T,T]

  // Workspace (f16): Qh | Kh | VhT | Oh  -> 4 x 4 MB = 16 MB
  constexpr size_t kQK = (size_t)kB * kH * kT * kDK;  // 2,097,152 halves
  _Float16* wsh = (_Float16*)d_ws;
  _Float16* Qh  = wsh;
  _Float16* Kh  = wsh + kQK;
  _Float16* VhT = wsh + 2 * kQK;
  _Float16* Oh  = wsh + 3 * kQK;

  const dim3 blk(32, 4);  // 4 independent waves per workgroup
  constexpr int gProj = ((kB * kT / 16) * (kD / 64)) / 4;  // 512
  constexpr int gAttn = (kB * kH * (kT / 16)) / 4;         // 512

  mha_proj<<<gProj, blk, 0, stream>>>(q, Wq, bq, Qh, 0);
  mha_proj<<<gProj, blk, 0, stream>>>(k, Wk, bk, Kh, 1);
  mha_proj<<<gProj, blk, 0, stream>>>(v, Wv, bv, VhT, 2);
  mha_attn<<<gAttn, blk, 0, stream>>>(Qh, Kh, VhT, Oh, Aout);
  mha_outproj<<<gProj, blk, 0, stream>>>(Oh, Wo, bo, out);
}
